// ALiBiMultiHeadAttention_20340965114514
// MI455X (gfx1250) — compile-verified, matrix operand order checked by reading, not run on hardware
//
#include <hip/hip_runtime.h>

typedef __attribute__((ext_vector_type(16))) __bf16 v16bf;
typedef __attribute__((ext_vector_type(4)))  __bf16 v4bf;
typedef __attribute__((ext_vector_type(8)))  float  v8f;

#define S_LEN  4096
#define DMODEL 1024
#define NHEADS 16
#define HDIM   64
#define WINDOW 512
#define NEGM   (-1e30f)

union FragBF { v16bf v; uint4 u[2]; };

// Assemble a 16-element bf16 WMMA operand fragment from two contiguous
// 16-byte chunks (elements 0..7 and 8..15 of the lane's register slice).
__device__ inline v16bf load_frag(const __bf16* p0, const __bf16* p1) {
  FragBF f;
  f.u[0] = *(const uint4*)p0;
  f.u[1] = *(const uint4*)p1;
  return f.v;
}

// ---------------- fp32 -> bf16 convert for X ----------------
__global__ void cvt_x(const float* __restrict__ x, __bf16* __restrict__ xb, int n4) {
  int i = blockIdx.x * blockDim.x + threadIdx.x;
  if (i < n4) {
    float4 f = ((const float4*)x)[i];
    v4bf r;
    r[0] = (__bf16)f.x; r[1] = (__bf16)f.y; r[2] = (__bf16)f.z; r[3] = (__bf16)f.w;
    ((v4bf*)xb)[i] = r;
  }
}

// ------------- fp32 -> bf16 convert + transpose for W (to W^T[n][k]) -------------
__global__ void cvt_wt(const float* __restrict__ w, __bf16* __restrict__ wt, int total) {
  int i = blockIdx.x * blockDim.x + threadIdx.x;   // i = k*3072 + n  (coalesced read)
  if (i < total) {
    int k = i / (3 * DMODEL);
    int n = i % (3 * DMODEL);
    wt[(size_t)n * DMODEL + k] = (__bf16)w[i];
  }
}

// ---------------- QKV projection GEMM: one wave -> 32x64 tile ----------------
// Register-blocked 2x4 subtiles of 16x16: each A fragment reused 4x, each B
// fragment reused 2x (1.5 128b loads per WMMA instead of 4). Epilogue scatters
// to K[h][s][d], Q[h][s][d], and transposed V^T[h][d][s] (all bf16).
__global__ __launch_bounds__(256) void qkv_gemm(const __bf16* __restrict__ xb,
                                                const __bf16* __restrict__ wt,
                                                __bf16* __restrict__ kmat,
                                                __bf16* __restrict__ qmat,
                                                __bf16* __restrict__ vtm) {
  int wave = blockIdx.x * (blockDim.x >> 5) + (threadIdx.x >> 5);
  int lane = threadIdx.x & 31;
  int row  = lane & 15;        // M/N index inside a 16x16 subtile
  int hl   = lane >> 4;        // lane half selects K sub-chunk

  int mg = wave / 48;          // 128 row groups of 32
  int cg = wave % 48;          // 48 col groups of 64
  int row0 = mg * 32, col0 = cg * 64;

  const __bf16* arow0 = xb + (size_t)(row0 + row) * DMODEL;
  const __bf16* arow1 = xb + (size_t)(row0 + 16 + row) * DMODEL;
  const __bf16* brow[4];
#pragma unroll
  for (int nt = 0; nt < 4; ++nt)
    brow[nt] = wt + (size_t)(col0 + nt * 16 + row) * DMODEL;

  v8f acc[2][4] = {};
#pragma unroll 2
  for (int k = 0; k < DMODEL; k += 32) {
    v16bf a0 = load_frag(arow0 + k + 8 * hl, arow0 + k + 16 + 8 * hl);
    v16bf a1 = load_frag(arow1 + k + 8 * hl, arow1 + k + 16 + 8 * hl);
#pragma unroll
    for (int nt = 0; nt < 4; ++nt) {
      v16bf b = load_frag(brow[nt] + k + 8 * hl, brow[nt] + k + 16 + 8 * hl);
      acc[0][nt] = __builtin_amdgcn_wmma_f32_16x16x32_bf16(false, a0, false, b, (short)0, acc[0][nt], false, false);
      acc[1][nt] = __builtin_amdgcn_wmma_f32_16x16x32_bf16(false, a1, false, b, (short)0, acc[1][nt], false, false);
    }
  }

  // Epilogue. The 64-column group lies entirely inside one head of one of
  // K/Q/V (col0 is a multiple of 64). C layout: lane holds rows (r + 8*hl)
  // of each subtile, fixed column (16*nt + row) within the group.
  int seg = col0 >> 10;                 // 0=K, 1=Q, 2=V
  int h   = (col0 & 1023) >> 6;
  if (seg < 2) {
    __bf16* base = (seg ? qmat : kmat) + (size_t)h * S_LEN * HDIM;
#pragma unroll
    for (int mt = 0; mt < 2; ++mt)
#pragma unroll
      for (int nt = 0; nt < 4; ++nt)
#pragma unroll
        for (int r = 0; r < 8; ++r) {
          int s = row0 + mt * 16 + 8 * hl + r;
          base[(size_t)s * HDIM + nt * 16 + row] = (__bf16)acc[mt][nt][r];
        }
  } else {
    __bf16* base = vtm + (size_t)h * HDIM * S_LEN;
#pragma unroll
    for (int mt = 0; mt < 2; ++mt)
#pragma unroll
      for (int nt = 0; nt < 4; ++nt)
#pragma unroll
        for (int r = 0; r < 8; ++r) {
          int s = row0 + mt * 16 + 8 * hl + r;
          base[(size_t)(nt * 16 + row) * S_LEN + s] = (__bf16)acc[mt][nt][r];
        }
  }
}

__device__ inline float rmax16(float v) {
  v = fmaxf(v, __shfl_xor(v, 1, 32));
  v = fmaxf(v, __shfl_xor(v, 2, 32));
  v = fmaxf(v, __shfl_xor(v, 4, 32));
  v = fmaxf(v, __shfl_xor(v, 8, 32));
  return v;
}
__device__ inline float rsum16(float v) {
  v += __shfl_xor(v, 1, 32);
  v += __shfl_xor(v, 2, 32);
  v += __shfl_xor(v, 4, 32);
  v += __shfl_xor(v, 8, 32);
  return v;
}

// ---------------- Flash-style sliding-window ALiBi attention ----------------
// One wave per (head, 16-query tile). 32 keys per step: 4 WMMAs for S=QK^T,
// online softmax in registers, P bounced through LDS (C-layout -> A-layout),
// 4 WMMAs for O += P*V.
__global__ __launch_bounds__(128) void attn(const __bf16* __restrict__ kmat,
                                            const __bf16* __restrict__ qmat,
                                            const __bf16* __restrict__ vtm,
                                            float* __restrict__ out) {
  __shared__ __bf16 plds[4][16 * 32];

  int wv   = threadIdx.x >> 5;
  int wave = blockIdx.x * 4 + wv;
  int h    = wave >> 8;           // 256 query tiles per head
  int qt   = wave & 255;
  int qb0  = qt * 16;
  int lane = threadIdx.x & 31;
  int row  = lane & 15;
  int hl   = lane >> 4;

  float slope = exp2f(-0.5f * (float)(h + 1));   // 1/(2^8)^((h+1)/16)
  const float scale = 1.0f / 32.0f;              // 1/sqrt(1024)

  // Q fragments (A-matrix, 16 rows x 64 dims = two K-chunks), loop invariant.
  const __bf16* qrow = qmat + ((size_t)(h * S_LEN + qb0 + row)) * HDIM;
  v16bf aq0 = load_frag(qrow + 8 * hl,      qrow + 16 + 8 * hl);
  v16bf aq1 = load_frag(qrow + 32 + 8 * hl, qrow + 48 + 8 * hl);

  float m[8], l[8];
#pragma unroll
  for (int r = 0; r < 8; ++r) { m[r] = NEGM; l[r] = 0.0f; }
  v8f oacc[4] = {};

  const __bf16* khead = kmat + (size_t)h * S_LEN * HDIM;
  const __bf16* vhead = vtm  + (size_t)h * HDIM * S_LEN;

  int kstart = (qb0 >= WINDOW) ? ((qb0 - WINDOW) & ~31) : 0;
  for (int kbp = kstart; kbp < qb0 + 16; kbp += 32) {
    // Prefetch next step's K rows / V columns (global_prefetch_b8).
    int nk = kbp + 32;
    if (nk < qb0 + 16) {
      __builtin_prefetch(khead + (size_t)(nk + row) * HDIM, 0, 0);
      __builtin_prefetch(vhead + (size_t)row * S_LEN + nk, 0, 0);
    }

    // K fragments: B-operand rows are K rows (B = K^T of the score GEMM).
    const __bf16* k0r = khead + (size_t)(kbp + row) * HDIM;
    const __bf16* k1r = khead + (size_t)(kbp + 16 + row) * HDIM;
    v16bf b00 = load_frag(k0r + 8 * hl,      k0r + 16 + 8 * hl);
    v16bf b01 = load_frag(k0r + 32 + 8 * hl, k0r + 48 + 8 * hl);
    v16bf b10 = load_frag(k1r + 8 * hl,      k1r + 16 + 8 * hl);
    v16bf b11 = load_frag(k1r + 32 + 8 * hl, k1r + 48 + 8 * hl);

    v8f s0 = {}, s1 = {};
    s0 = __builtin_amdgcn_wmma_f32_16x16x32_bf16(false, aq0, false, b00, (short)0, s0, false, false);
    s0 = __builtin_amdgcn_wmma_f32_16x16x32_bf16(false, aq1, false, b01, (short)0, s0, false, false);
    s1 = __builtin_amdgcn_wmma_f32_16x16x32_bf16(false, aq0, false, b10, (short)0, s1, false, false);
    s1 = __builtin_amdgcn_wmma_f32_16x16x32_bf16(false, aq1, false, b11, (short)0, s1, false, false);

    int j0 = kbp + row, j1 = kbp + 16 + row;
#pragma unroll
    for (int r = 0; r < 8; ++r) {
      int i  = qb0 + r + 8 * hl;
      int d0 = i - j0, d1 = i - j1;
      float v0 = (d0 >= 0 && d0 <= WINDOW) ? (s0[r] * scale - slope * (float)d0) : NEGM;
      float v1 = (d1 >= 0 && d1 <= WINDOW) ? (s1[r] * scale - slope * (float)d1) : NEGM;

      float rm = rmax16(fmaxf(v0, v1));
      float mn = fmaxf(m[r], rm);
      float alpha = __expf(m[r] - mn);
      float p0 = __expf(v0 - mn);
      float p1 = __expf(v1 - mn);
      float ps = rsum16(p0 + p1);
      l[r] = l[r] * alpha + ps;
      m[r] = mn;
#pragma unroll
      for (int t = 0; t < 4; ++t) oacc[t][r] *= alpha;

      // P in C-layout -> LDS (row-major 16x32 bf16 tile)
      plds[wv][(r + 8 * hl) * 32 + row]      = (__bf16)p0;
      plds[wv][(r + 8 * hl) * 32 + 16 + row] = (__bf16)p1;
    }

    // Same-wave LDS store->load ordering (cross-lane): drain DS counter and
    // stop the compiler from reordering across the relayout.
    asm volatile("s_wait_dscnt 0" ::: "memory");

    // P in A-layout: lane row = query row, two contiguous 16B K-chunks.
    const __bf16* prow = &plds[wv][row * 32];
    v16bf ap = load_frag(prow + 8 * hl, prow + 16 + 8 * hl);

    // O += P * V  (B-operand rows are V^T rows = dim rows, contiguous in s)
#pragma unroll
    for (int t = 0; t < 4; ++t) {
      const __bf16* vr = vhead + (size_t)(t * 16 + row) * S_LEN + kbp;
      v16bf bv = load_frag(vr + 8 * hl, vr + 16 + 8 * hl);
      oacc[t] = __builtin_amdgcn_wmma_f32_16x16x32_bf16(false, ap, false, bv, (short)0, oacc[t], false, false);
    }
    asm volatile("" ::: "memory");  // keep LDS reads of this iter before next iter's writes
  }

  // Normalize and store: out[s][h*64 + d], C-layout rows r+8*hl, col row per dim tile.
  float* orow = out + (size_t)(qb0 + 8 * hl) * DMODEL + h * HDIM + row;
#pragma unroll
  for (int r = 0; r < 8; ++r) {
    float inv = 1.0f / l[r];
#pragma unroll
    for (int t = 0; t < 4; ++t)
      orow[(size_t)r * DMODEL + t * 16] = oacc[t][r] * inv;
  }
}

extern "C" void kernel_launch(void* const* d_in, const int* in_sizes, int n_in,
                              void* d_out, int out_size, void* d_ws, size_t ws_size,
                              hipStream_t stream) {
  const float* x = (const float*)d_in[0];   // [1,4096,1024] fp32
  const float* w = (const float*)d_in[1];   // [1024,3072]  fp32
  float* out = (float*)d_out;               // [1,4096,1024] fp32
  char* ws = (char*)d_ws;

  // Workspace layout (bytes):
  //   Xb  bf16 [4096][1024]      @ 0        (8 MB)
  //   Wt  bf16 [3072][1024]      @ 8 MB     (6 MB)
  //   K   bf16 [16][4096][64]    @ 14 MB    (8 MB)
  //   Q   bf16 [16][4096][64]    @ 22 MB    (8 MB)
  //   Vt  bf16 [16][64][4096]    @ 30 MB    (8 MB)
  __bf16* xb = (__bf16*)(ws);
  __bf16* wt = (__bf16*)(ws + (size_t)8  * 1024 * 1024);
  __bf16* km = (__bf16*)(ws + (size_t)14 * 1024 * 1024);
  __bf16* qm = (__bf16*)(ws + (size_t)22 * 1024 * 1024);
  __bf16* vt = (__bf16*)(ws + (size_t)30 * 1024 * 1024);

  hipLaunchKernelGGL(cvt_x,    dim3(4096),  dim3(256), 0, stream, x, xb, S_LEN * DMODEL / 4);
  hipLaunchKernelGGL(cvt_wt,   dim3(12288), dim3(256), 0, stream, w, wt, DMODEL * 3 * DMODEL);
  // 128 row groups (32 rows) x 48 col groups (64 cols) = 6144 waves / 8 per block
  hipLaunchKernelGGL(qkv_gemm, dim3(768),   dim3(256), 0, stream, xb, wt, km, qm, vt);
  hipLaunchKernelGGL(attn,     dim3(1024),  dim3(128), 0, stream, km, qm, vt, out);
}
